// SpeculativeCrossLayerBlock_64141041598880
// MI455X (gfx1250) — matrix operand order repaired and checked by reading, not compiled, not run on hardware
//
#include <hip/hip_runtime.h>
#include <hip/hip_bf16.h>
#include <math.h>

typedef _Float16 half_t;
typedef __attribute__((ext_vector_type(16))) _Float16 v16h;
typedef __attribute__((ext_vector_type(8)))  float    v8f;

#define TT 32          // seq len
#define CC 128         // embed dim
#define NH 4           // heads
#define HS 32          // head size
#define NE 8           // experts
#define HF 512         // expert hidden
#define NB 4096        // batch
#define BT (NB * TT)   // tokens = 131072
#define NEG_INF_F (-1e30f)

// ---------------------------------------------------------------------------
// WMMA helpers (CDNA5 v_wmma_f32_16x16x32_f16, wave32)
// ---------------------------------------------------------------------------
__device__ __forceinline__ v8f wmma16(v16h a, v16h b, v8f c) {
  return __builtin_amdgcn_wmma_f32_16x16x32_f16(
      /*neg_a=*/false, a, /*neg_b=*/false, b,
      /*c_mod=*/(short)0, c, /*reuse_a=*/false, /*reuse_b=*/false);
}

union V16HU { struct { uint4 lo, hi; } q; v16h v; };
union H8U   { uint4 u; half_t h[8]; };

// A fragment (16x32 f16, row-major tile, row stride ldh halves), two b128 loads.
// Lanes 0-15: M=lane, halves 0-7 -> K=0..7, halves 8-15 -> K=16..23;
// lanes 16-31 shift K by +8.  (16-byte alignment guaranteed by callers.)
__device__ __forceinline__ v16h ld_A(const half_t* tile, int ldh) {
  const int lane = threadIdx.x & 31;
  const half_t* r = tile + (lane & 15) * ldh + ((lane >> 4) << 3);
  V16HU cv;
  cv.q.lo = *(const uint4*)r;
  cv.q.hi = *(const uint4*)(r + 16);
  return cv.v;
}

// B fragment from N-major storage Bt[n][k]: lanes 0-15 N=lane, halves 0-7 ->
// K=0..7, halves 8-15 -> K=16..23; lanes 16-31 shift K by +8.
__device__ __forceinline__ v16h ld_Bt(const half_t* tile, int ldh) {
  const int lane = threadIdx.x & 31;
  const half_t* r = tile + (lane & 15) * ldh + ((lane >> 4) << 3);
  V16HU cv;
  cv.q.lo = *(const uint4*)r;
  cv.q.hi = *(const uint4*)(r + 16);
  return cv.v;
}

// wave32 reductions
__device__ __forceinline__ float wave_sum32(float v) {
#pragma unroll
  for (int o = 16; o > 0; o >>= 1) v += __shfl_xor(v, o, 32);
  return v;
}
// reductions inside each 16-lane half (rows r / r+8 of a C/D tile)
__device__ __forceinline__ float sum16(float v) {
#pragma unroll
  for (int o = 8; o > 0; o >>= 1) v += __shfl_xor(v, o, 32);
  return v;
}
__device__ __forceinline__ float max16(float v) {
#pragma unroll
  for (int o = 8; o > 0; o >>= 1) v = fmaxf(v, __shfl_xor(v, o, 32));
  return v;
}

// ---------------------------------------------------------------------------
// Prep: f32 [batch][R][Cc] -> f16 [batch][Cc][R] (N-major for WMMA B operand)
// ---------------------------------------------------------------------------
__global__ __launch_bounds__(256) void transpose_f16_kernel(
    const float* __restrict__ in, half_t* __restrict__ out, int R, int Cc) {
  const size_t bo = (size_t)blockIdx.y * R * Cc;
  for (int idx = blockIdx.x * 256 + threadIdx.x; idx < R * Cc;
       idx += gridDim.x * 256) {
    const int r = idx / Cc, c = idx % Cc;
    out[bo + (size_t)c * R + r] = (half_t)in[bo + idx];
  }
}

__global__ void zero_cnt_kernel(int* cnt) {
  if (threadIdx.x < NE) cnt[threadIdx.x] = 0;
}

// ---------------------------------------------------------------------------
// Kernel 1: LN1 + QKV + causal attention + proj + residual. 1 block per batch b.
// ---------------------------------------------------------------------------
__global__ __launch_bounds__(256) void attn_kernel(
    const float* __restrict__ x, const float* __restrict__ ln1w,
    const float* __restrict__ ln1b, const half_t* __restrict__ WqT,
    const half_t* __restrict__ WkT, const half_t* __restrict__ WvT,
    const half_t* __restrict__ WoT, const float* __restrict__ bo,
    float* __restrict__ x1) {
  __shared__ __align__(16) float  xs[TT][CC];      // original x (residual)
  __shared__ __align__(16) half_t hln[TT][CC];     // LN1(x), A for QKV
  __shared__ __align__(16) half_t qs[NH][TT][HS];  // q, A for scores
  __shared__ __align__(16) half_t ks[NH][TT][HS];  // k [u][s]: Bt for scores
  __shared__ __align__(16) half_t vt[NH][HS][TT];  // v^T [s][u]: Bt for attn@v
  __shared__ __align__(16) half_t wei[NH][TT][TT]; // softmax, A for attn@v
  __shared__ __align__(16) half_t ao[TT][CC];      // attn out, A for proj

  const int tid = threadIdx.x, lane = tid & 31;
  const int wave = __builtin_amdgcn_readfirstlane(tid >> 5);  // scalar!
  const size_t bbase = (size_t)blockIdx.x * TT * CC;

  for (int i = tid; i < TT * CC; i += 256) (&xs[0][0])[i] = x[bbase + i];
  __syncthreads();

  // LN1: 8 waves x 4 rows, lane covers 4 contiguous channels
  for (int i = 0; i < 4; ++i) {
    const int t = wave * 4 + i;
    float v0[4], s = 0.f;
#pragma unroll
    for (int j = 0; j < 4; ++j) { v0[j] = xs[t][lane * 4 + j]; s += v0[j]; }
    s = wave_sum32(s);
    const float mu = s * (1.f / CC);
    float var = 0.f;
#pragma unroll
    for (int j = 0; j < 4; ++j) { float d = v0[j] - mu; var += d * d; }
    var = wave_sum32(var);
    const float inv = rsqrtf(var * (1.f / CC) + 1e-5f);
    union { uint2 u; half_t h[4]; } pk;
#pragma unroll
    for (int j = 0; j < 4; ++j) {
      const int c = lane * 4 + j;
      pk.h[j] = (half_t)((v0[j] - mu) * inv * ln1w[c] + ln1b[c]);
    }
    *(uint2*)&hln[t][lane * 4] = pk.u;  // 8B packed DS store
  }
  __syncthreads();

  // Phase A: waves 0-3 -> Q(head=wave), waves 4-7 -> K(head=wave-4).
  // Q and K share the [token][s] row-major store layout -> branch-free body.
  {
    const int head = wave & 3;
    const half_t* WT = (wave < 4 ? WqT : WkT) + head * HS * CC;
    half_t* dst = (wave < 4 ? &qs[head][0][0] : &ks[head][0][0]);
#pragma unroll
    for (int mt = 0; mt < 2; ++mt)
#pragma unroll
      for (int nt = 0; nt < 2; ++nt) {
        v8f acc = {};
#pragma unroll
        for (int kk = 0; kk < 4; ++kk)
          acc = wmma16(ld_A(&hln[mt * 16][kk * 32], CC),
                       ld_Bt(WT + nt * 16 * CC + kk * 32, CC), acc);
        const int n = nt * 16 + (lane & 15);
        const int mb = mt * 16 + ((lane >> 4) << 3);
#pragma unroll
        for (int r = 0; r < 8; ++r) dst[(mb + r) * HS + n] = (half_t)acc[r];
      }
  }
  // Phase B: waves 0-3 -> V(head=wave), stored transposed; the 8 accumulator
  // rows are m-consecutive -> one b128 DS store per tile.
  if (wave < 4) {
    const int head = wave;
    const half_t* WT = WvT + head * HS * CC;
#pragma unroll
    for (int mt = 0; mt < 2; ++mt)
#pragma unroll
      for (int nt = 0; nt < 2; ++nt) {
        v8f acc = {};
#pragma unroll
        for (int kk = 0; kk < 4; ++kk)
          acc = wmma16(ld_A(&hln[mt * 16][kk * 32], CC),
                       ld_Bt(WT + nt * 16 * CC + kk * 32, CC), acc);
        const int n = nt * 16 + (lane & 15);       // s index (row of vt)
        const int mb = mt * 16 + ((lane >> 4) << 3);
        H8U pk;
#pragma unroll
        for (int r = 0; r < 8; ++r) pk.h[r] = (half_t)acc[r];
        *(uint4*)&vt[head][n][mb] = pk.u;
      }
  }
  __syncthreads();

  // Scores + causal softmax + attn@v: wave -> (head = wave>>1, mt = wave&1)
  {
    const int head = wave >> 1, mt = wave & 1;
    v8f c0 = {}, c1 = {};
    const v16h aq = ld_A(&qs[head][mt * 16][0], HS);   // K = 32, one step
    c0 = wmma16(aq, ld_Bt(&ks[head][0][0], HS), c0);
    c1 = wmma16(aq, ld_Bt(&ks[head][16][0], HS), c1);
    const float scale = 0.08838834764831845f;          // C^-0.5 (full embed)
#pragma unroll
    for (int r = 0; r < 8; ++r) {
      const int m = mt * 16 + r + ((lane >> 4) << 3);
      const int n0 = lane & 15, n1 = n0 + 16;
      const float s0 = (n0 <= m) ? c0[r] * scale : NEG_INF_F;
      const float s1 = (n1 <= m) ? c1[r] * scale : NEG_INF_F;
      const float mx = max16(fmaxf(s0, s1));
      const float e0 = __expf(s0 - mx), e1 = __expf(s1 - mx);
      const float inv = 1.f / sum16(e0 + e1);
      wei[head][m][n0] = (half_t)(e0 * inv);
      wei[head][m][n1] = (half_t)(e1 * inv);
    }
    __syncthreads();
    v8f d0 = {}, d1 = {};
    const v16h aw = ld_A(&wei[head][mt * 16][0], TT);  // K = 32 tokens
    d0 = wmma16(aw, ld_Bt(&vt[head][0][0], TT), d0);
    d1 = wmma16(aw, ld_Bt(&vt[head][16][0], TT), d1);
    const int co = head * HS + (lane & 15);
    const int mb = mt * 16 + ((lane >> 4) << 3);
#pragma unroll
    for (int r = 0; r < 8; ++r) {
      ao[mb + r][co] = (half_t)d0[r];
      ao[mb + r][co + 16] = (half_t)d1[r];
    }
  }
  __syncthreads();

  // Proj (32x128x128) + bias + residual; wave owns one 16-wide N slice
  {
    const int n0 = wave * 16;
#pragma unroll
    for (int mt = 0; mt < 2; ++mt) {
      v8f acc = {};
#pragma unroll
      for (int kk = 0; kk < 4; ++kk)
        acc = wmma16(ld_A(&ao[mt * 16][kk * 32], CC),
                     ld_Bt(WoT + n0 * CC + kk * 32, CC), acc);
      const int n = n0 + (lane & 15);
      const int mb = mt * 16 + ((lane >> 4) << 3);
      const float bn_ = bo[n];
#pragma unroll
      for (int r = 0; r < 8; ++r)
        x1[bbase + (mb + r) * CC + n] = acc[r] + bn_ + xs[mb + r][n];
    }
  }
}

// ---------------------------------------------------------------------------
// Kernel 2: LN2 + noisy router + top-2 dispatch. One wave per token.
// ---------------------------------------------------------------------------
__global__ __launch_bounds__(256) void router_kernel(
    const float* __restrict__ x1, const float* __restrict__ noise,
    const float* __restrict__ ln2w, const float* __restrict__ ln2b,
    const float* __restrict__ Wr, const float* __restrict__ br,
    const float* __restrict__ Wn, const float* __restrict__ bn,
    const float* __restrict__ Wsk, const float* __restrict__ bsk,
    half_t* __restrict__ nxh, int* __restrict__ cnt, int* __restrict__ list,
    float* __restrict__ wlist, unsigned char* __restrict__ nonskip) {
  const int lane = threadIdx.x & 31;
  const int wave = __builtin_amdgcn_readfirstlane(threadIdx.x >> 5);
  const int t = blockIdx.x * 8 + wave;

  const float4 xv = *(const float4*)(x1 + (size_t)t * CC + lane * 4);
  const float v0[4] = {xv.x, xv.y, xv.z, xv.w};
  float s = v0[0] + v0[1] + v0[2] + v0[3];
  s = wave_sum32(s);
  const float mu = s * (1.f / CC);
  float var = 0.f;
#pragma unroll
  for (int j = 0; j < 4; ++j) { float d = v0[j] - mu; var += d * d; }
  var = wave_sum32(var);
  const float inv = rsqrtf(var * (1.f / CC) + 1e-5f);
  float nx[4];
  union { uint2 u; half_t h[4]; } pk;
#pragma unroll
  for (int j = 0; j < 4; ++j) {
    const int c = lane * 4 + j;
    nx[j] = (v0[j] - mu) * inv * ln2w[c] + ln2b[c];
    pk.h[j] = (half_t)nx[j];
  }
  *(uint2*)(nxh + (size_t)t * CC + lane * 4) = pk.u;

  // router / noise logits: each c-row of Wr/Wn is 8 contiguous floats
  float pl[NE] = {}, pn[NE] = {};
#pragma unroll
  for (int j = 0; j < 4; ++j) {
    const int c = lane * 4 + j;
    const float4 r0 = *(const float4*)(Wr + c * NE);
    const float4 r1 = *(const float4*)(Wr + c * NE + 4);
    const float4 q0 = *(const float4*)(Wn + c * NE);
    const float4 q1 = *(const float4*)(Wn + c * NE + 4);
    pl[0] += nx[j] * r0.x; pl[1] += nx[j] * r0.y;
    pl[2] += nx[j] * r0.z; pl[3] += nx[j] * r0.w;
    pl[4] += nx[j] * r1.x; pl[5] += nx[j] * r1.y;
    pl[6] += nx[j] * r1.z; pl[7] += nx[j] * r1.w;
    pn[0] += nx[j] * q0.x; pn[1] += nx[j] * q0.y;
    pn[2] += nx[j] * q0.z; pn[3] += nx[j] * q0.w;
    pn[4] += nx[j] * q1.x; pn[5] += nx[j] * q1.y;
    pn[6] += nx[j] * q1.z; pn[7] += nx[j] * q1.w;
  }
  float noisy[NE];
#pragma unroll
  for (int e = 0; e < NE; ++e) {
    const float l = wave_sum32(pl[e]) + br[e];
    const float nl = wave_sum32(pn[e]) + bn[e];
    const float sp = (nl > 20.f) ? nl : log1pf(__expf(nl));  // softplus
    noisy[e] = l + noise[(size_t)t * NE + e] * sp;
  }
  const float4 wk = *(const float4*)(Wsk + lane * 4);
  float psk = nx[0] * wk.x + nx[1] * wk.y + nx[2] * wk.z + nx[3] * wk.w;
  psk = wave_sum32(psk) + bsk[0];
  const bool ns = (psk <= 0.f);  // sigmoid(psk) <= 0.5

  // top-2 with first-index tie-break (matches jax.lax.top_k)
  int e0 = 0; float b0 = noisy[0];
#pragma unroll
  for (int e = 1; e < NE; ++e) if (noisy[e] > b0) { b0 = noisy[e]; e0 = e; }
  int e1 = (e0 == 0) ? 1 : 0; float b1v = noisy[e1];
#pragma unroll
  for (int e = 0; e < NE; ++e)
    if (e != e0 && noisy[e] > b1v) { b1v = noisy[e]; e1 = e; }
  const float z = __expf(b1v - b0);            // softmax over top-2 support
  const float w0 = 1.f / (1.f + z), w1 = z / (1.f + z);

  if (lane == 0) {
    nonskip[t] = ns ? 1 : 0;
    if (ns) {
      const int p0 = atomicAdd(&cnt[e0], 1);
      list[e0 * BT + p0] = t * 2;     wlist[e0 * BT + p0] = w0;
      const int p1 = atomicAdd(&cnt[e1], 1);
      list[e1 * BT + p1] = t * 2 + 1; wlist[e1 * BT + p1] = w1;
    }
  }
}

// ---------------------------------------------------------------------------
// Kernel 3: per-expert gathered FFN. grid = (maxTiles, E); early-exit if empty.
// ---------------------------------------------------------------------------
__global__ __launch_bounds__(256) void expert_kernel(
    const half_t* __restrict__ nxh, const half_t* __restrict__ W1T,
    const float* __restrict__ b1, const half_t* __restrict__ W2T,
    const float* __restrict__ b2, const int* __restrict__ cnt,
    const int* __restrict__ list, const float* __restrict__ wlist,
    half_t* __restrict__ res) {
  const int e = blockIdx.y;
  const int base = blockIdx.x * 32;
  const int n_e = cnt[e];
  if (base >= n_e) return;
  const int m_valid = min(32, n_e - base);

  __shared__ __align__(16) half_t A[TT][CC];    // gathered tokens (A-matrix)
  __shared__ __align__(16) half_t act[TT][HF];  // relu(h@W1+b1), A for GEMM2
  __shared__ int toks[32];
  __shared__ float wts[32];
  __shared__ unsigned char vld[32];

  const int tid = threadIdx.x, lane = tid & 31;
  const int wave = __builtin_amdgcn_readfirstlane(tid >> 5);
  if (tid < 32) {
    if (tid < m_valid) {
      toks[tid] = list[e * BT + base + tid];
      wts[tid] = wlist[e * BT + base + tid];
      vld[tid] = 1;
    } else { toks[tid] = 0; wts[tid] = 0.f; vld[tid] = 0; }
  }
  __syncthreads();
  {  // gather 32 token rows of nxh into LDS, 32B per thread
    const int row = tid >> 3, ch = (tid & 7) * 16;
    const uint4* src = (const uint4*)(nxh + (size_t)(toks[row] >> 1) * CC + ch);
    uint4* dst = (uint4*)&A[row][ch];
    dst[0] = src[0];
    dst[1] = src[1];
  }
  __syncthreads();

  const half_t* W1e = W1T + (size_t)e * HF * CC;
  const half_t* W2e = W2T + (size_t)e * CC * HF;
  __builtin_prefetch(W1e + (size_t)wave * 64 * CC, 0, 1);
  __builtin_prefetch(W2e + (size_t)wave * 16 * HF, 0, 1);

  // GEMM1: 32 x 512 x 128, wave owns a 64-wide N slice
#pragma unroll
  for (int nt = 0; nt < 4; ++nt) {
    const int n0 = wave * 64 + nt * 16;
#pragma unroll
    for (int mt = 0; mt < 2; ++mt) {
      v8f acc = {};
#pragma unroll
      for (int kk = 0; kk < 4; ++kk)
        acc = wmma16(ld_A(&A[mt * 16][kk * 32], CC),
                     ld_Bt(W1e + n0 * CC + kk * 32, CC), acc);
      const int n = n0 + (lane & 15);
      const int mb = mt * 16 + ((lane >> 4) << 3);
      const float bias = b1[e * HF + n];
#pragma unroll
      for (int r = 0; r < 8; ++r)
        act[mb + r][n] = (half_t)fmaxf(acc[r] + bias, 0.f);
    }
  }
  __syncthreads();

  // GEMM2: 32 x 128 x 512, wave owns a 16-wide N slice; scatter weighted rows
  {
    const int n0 = wave * 16;
#pragma unroll
    for (int mt = 0; mt < 2; ++mt) {
      v8f acc = {};
#pragma unroll
      for (int kk = 0; kk < 16; ++kk)
        acc = wmma16(ld_A(&act[mt * 16][kk * 32], HF),
                     ld_Bt(W2e + n0 * HF + kk * 32, HF), acc);
      const int n = n0 + (lane & 15);
      const int mb = mt * 16 + ((lane >> 4) << 3);
      const float bias = b2[e * CC + n];
#pragma unroll
      for (int r = 0; r < 8; ++r) {
        const int m = mb + r;
        if (vld[m])
          res[(size_t)toks[m] * CC + n] = (half_t)((acc[r] + bias) * wts[m]);
      }
    }
  }
}

// ---------------------------------------------------------------------------
// Kernel 4: out = x1 + nonskip * (res[2t] + res[2t+1])
// ---------------------------------------------------------------------------
__global__ __launch_bounds__(256) void combine_kernel(
    float* __restrict__ out, const half_t* __restrict__ res,
    const unsigned char* __restrict__ nonskip) {
  const size_t idx = (size_t)blockIdx.x * 256 + threadIdx.x;
  const size_t t = idx / CC;
  const int c = (int)(idx % CC);
  float v = out[idx];
  if (nonskip[t])
    v += (float)res[(t * 2) * CC + c] + (float)res[(t * 2 + 1) * CC + c];
  out[idx] = v;
}

// ---------------------------------------------------------------------------
extern "C" void kernel_launch(void* const* d_in, const int* in_sizes, int n_in,
                              void* d_out, int out_size, void* d_ws,
                              size_t ws_size, hipStream_t stream) {
  const float* x     = (const float*)d_in[0];
  const float* noise = (const float*)d_in[1];
  const float* ln1w  = (const float*)d_in[2];
  const float* ln1b  = (const float*)d_in[3];
  const float* ln2w  = (const float*)d_in[4];
  const float* ln2b  = (const float*)d_in[5];
  const float* Wq    = (const float*)d_in[6];
  const float* Wk    = (const float*)d_in[7];
  const float* Wv    = (const float*)d_in[8];
  const float* Wo    = (const float*)d_in[9];
  const float* bo    = (const float*)d_in[10];
  const float* Wr    = (const float*)d_in[11];
  const float* br    = (const float*)d_in[12];
  const float* Wn    = (const float*)d_in[13];
  const float* bn    = (const float*)d_in[14];
  const float* Wsk   = (const float*)d_in[15];
  const float* bsk   = (const float*)d_in[16];
  const float* W1    = (const float*)d_in[17];
  const float* b1    = (const float*)d_in[18];
  const float* W2    = (const float*)d_in[19];
  const float* b2    = (const float*)d_in[20];
  float* out = (float*)d_out;

  // workspace carve-up (256B aligned), ~110 MB total
  char* ws = (char*)d_ws;
  size_t off = 0;
  auto carve = [&](size_t bytes) {
    char* p = ws + off;
    off = (off + bytes + 255) & ~(size_t)255;
    return p;
  };
  half_t* nxh   = (half_t*)carve((size_t)BT * CC * 2);        // 32 MB
  half_t* res   = (half_t*)carve((size_t)2 * BT * CC * 2);    // 64 MB
  half_t* WqT   = (half_t*)carve((size_t)NH * HS * CC * 2);
  half_t* WkT   = (half_t*)carve((size_t)NH * HS * CC * 2);
  half_t* WvT   = (half_t*)carve((size_t)NH * HS * CC * 2);
  half_t* WoT   = (half_t*)carve((size_t)CC * CC * 2);
  half_t* W1T   = (half_t*)carve((size_t)NE * HF * CC * 2);   // 1 MB
  half_t* W2T   = (half_t*)carve((size_t)NE * CC * HF * 2);   // 1 MB
  int*    cnt   = (int*)carve(NE * sizeof(int));
  int*    list  = (int*)carve((size_t)NE * BT * sizeof(int)); // 4 MB
  float*  wlist = (float*)carve((size_t)NE * BT * sizeof(float));
  unsigned char* nonskip = (unsigned char*)carve((size_t)BT);
  (void)ws_size; (void)in_sizes; (void)n_in; (void)out_size;

  // prep: f16 N-major weight copies
  transpose_f16_kernel<<<dim3(16, NH), 256, 0, stream>>>(Wq, WqT, CC, HS);
  transpose_f16_kernel<<<dim3(16, NH), 256, 0, stream>>>(Wk, WkT, CC, HS);
  transpose_f16_kernel<<<dim3(16, NH), 256, 0, stream>>>(Wv, WvT, CC, HS);
  transpose_f16_kernel<<<dim3(64, 1), 256, 0, stream>>>(Wo, WoT, CC, CC);
  transpose_f16_kernel<<<dim3(256, NE), 256, 0, stream>>>(W1, W1T, CC, HF);
  transpose_f16_kernel<<<dim3(256, NE), 256, 0, stream>>>(W2, W2T, HF, CC);
  zero_cnt_kernel<<<1, 32, 0, stream>>>(cnt);

  attn_kernel<<<NB, 256, 0, stream>>>(x, ln1w, ln1b, WqT, WkT, WvT, WoT, bo,
                                      out);
  router_kernel<<<BT / 8, 256, 0, stream>>>(out, noise, ln2w, ln2b, Wr, br, Wn,
                                            bn, Wsk, bsk, nxh, cnt, list,
                                            wlist, nonskip);
  expert_kernel<<<dim3(BT / 32, NE), 256, 0, stream>>>(nxh, W1T, b1, W2T, b2,
                                                       cnt, list, wlist, res);
  combine_kernel<<<(size_t)BT * CC / 256, 256, 0, stream>>>(out, res, nonskip);
}
